// LiquidGaitObserver_2121713844985
// MI455X (gfx1250) — matrix operand order repaired and checked by reading, not compiled, not run on hardware
//
#include <hip/hip_runtime.h>

typedef __attribute__((ext_vector_type(16))) _Float16 v16h;
typedef __attribute__((ext_vector_type(2)))  __fp16   v2h;   // matches cvt_pkrtz return type
typedef __attribute__((ext_vector_type(8)))  float    v8f;
typedef __attribute__((ext_vector_type(8)))  unsigned v8u;

#define LOG2E 1.4426950408889634f

__device__ __forceinline__ float tanh_fast(float x) {
#if __has_builtin(__builtin_amdgcn_tanhf)
    return __builtin_amdgcn_tanhf(x);            // v_tanh_f32 (gfx1250 trans op)
#elif __has_builtin(__builtin_amdgcn_tanh_f32)
    return __builtin_amdgcn_tanh_f32(x);
#else
    // tanh(x) = 1 - 2/(exp(2x)+1); exact at saturation, uses v_exp/v_rcp trans units
    float e = __builtin_amdgcn_exp2f(x * (2.0f * LOG2E));
    float r = __builtin_amdgcn_rcpf(e + 1.0f);
    return fmaf(-2.0f, r, 1.0f);
#endif
}

// pack two f32 into one dword of two f16 (v_cvt_pk_rtz_f16_f32)
__device__ __forceinline__ unsigned pkU(float a, float b) {
    v2h p = __builtin_amdgcn_cvt_pkrtz(a, b);
    return __builtin_bit_cast(unsigned, p);
}

__device__ __forceinline__ void pk2(v16h& d, int e, float a, float b) {
    v2h p = __builtin_amdgcn_cvt_pkrtz(a, b);
    d[e]     = (_Float16)p[0];
    d[e + 1] = (_Float16)p[1];
}

// lane^16 exchange as a pure-VALU op: v_permlanex16_b32 with identity selects
// (each lane reads the same lane index in the opposite 16-lane row).
__device__ __forceinline__ unsigned xhalf(unsigned v) {
    return (unsigned)__builtin_amdgcn_permlanex16(
        (int)v, (int)v, 0x76543210, 0xfedcba98, false, false);
}

// Build a 32x16 f16 B fragment (K = 32 rows of h^T) from two 16x16 f32 D tiles.
// D layout: VGPR r holds row (tile*16 + r + 8*hi), col = lane%16.
// B order used here (contiguous K, as Arec): lanes 0-15 hold K=frag*32+0..15, lanes 16-31 hold K=frag*32+16..31.
// Pack f32->f16 pairs first; each half only needs ONE dword from the other half
// per j, so pre-select the outgoing dword and do a single permlanex16:
// per j: 2 pack + 1 permute + 3 cndmask, all VALU, no DS-unit traffic.
__device__ __forceinline__ v16h build_bfrag(const v8f Glo, const v8f Ghi, int hi) {
    v8u d;
#pragma unroll
    for (int j = 0; j < 4; ++j) {
        unsigned p0 = pkU(Glo[2 * j], Glo[2 * j + 1]);  // my tile-even rows 2j,2j+1
        unsigned p1 = pkU(Ghi[2 * j], Ghi[2 * j + 1]);  // my tile-odd  rows 2j,2j+1
        unsigned z  = hi ? p0 : p1;   // dword the OTHER half needs from me
        unsigned zz = xhalf(z);       // lo lanes get hi's p0; hi lanes get lo's p1
        d[j]     = hi ? zz : p0;      // elems 0..7  : rows 0..7  of my K-tile
        d[4 + j] = hi ? p1 : zz;      // elems 8..15 : rows 8..15 of my K-tile
    }
    return __builtin_bit_cast(v16h, d);
}

__global__ __launch_bounds__(256, 1)
void liquid_gait_kernel(const float* __restrict__ x,
                        const float* __restrict__ dtp,
                        const float* __restrict__ Win,
                        const float* __restrict__ bin,
                        const float* __restrict__ Wrec,
                        const float* __restrict__ brec,
                        const float* __restrict__ tau,
                        const float* __restrict__ Wcls,
                        const float* __restrict__ bcls,
                        const int* __restrict__ osp,
                        float* __restrict__ out,
                        int Bsz) {
    constexpr int T = 512, F = 6, H = 64;
    const int tid  = threadIdx.x;
    const int lane = tid & 31;
    const int wid  = (blockIdx.x * blockDim.x + tid) >> 5;
    const int col  = lane & 15;   // batch column (N) / A-row (M)
    const int hi   = lane >> 4;   // lane half
    const int b    = wid * 16 + col;

    // LDS: packed W_in|bias A-fragment dwords (4 frags x 16 cols x 4 dwords) + W_cls
    __shared__ unsigned lds_ain[4 * 16 * 4];
    __shared__ float    lds_wcls[64];

    if (tid < 64) {
        lds_wcls[tid] = Wcls[tid];
        int mt = tid >> 4, c = tid & 15;
        int m = mt * 16 + c;
        const float* wp = Win + m * F;
        unsigned* q = lds_ain + tid * 4;
        q[0] = pkU(wp[0], wp[1]);
        q[1] = pkU(wp[2], wp[3]);
        q[2] = pkU(wp[4], wp[5]);
        q[3] = pkU(bin[m] + brec[m], 0.0f);  // bias folded at K=6, K=7=0
    }
    __syncthreads();
    if (b >= Bsz) return;

    const int   OS  = osp[0];
    const float sdt = dtp[0] / (float)OS;
    const float bc  = bcls[0];

    // ---- A fragments of W_rec (16x16x32 f16 A layout), register-resident ----
    // contiguous K order used here (B matches): lane = M (both halves); lanes 0-15: elems 0..15 = K 0..15;
    //           lanes 16-31: elems 0..15 = K 16..31.
    v16h Arec[4][2];
#pragma unroll
    for (int mt = 0; mt < 4; ++mt) {
        int m = mt * 16 + col;
#pragma unroll
        for (int kf = 0; kf < 2; ++kf) {
            const float* wp = Wrec + m * H + kf * 32 + hi * 16;
            v16h a;
#pragma unroll
            for (int j = 0; j < 4; ++j) pk2(a, 2 * j, wp[2 * j], wp[2 * j + 1]);
#pragma unroll
            for (int j = 0; j < 4; ++j) pk2(a, 8 + 2 * j, wp[8 + 2 * j], wp[8 + 2 * j + 1]);
            Arec[mt][kf] = a;
        }
    }

    // ---- Per-lane decay in D layout: row m = mt*16 + hi*8 + r ----
    v8f dec8[4];
#pragma unroll
    for (int mt = 0; mt < 4; ++mt) {
        const float* tp = tau + mt * 16 + hi * 8;
#pragma unroll
        for (int r = 0; r < 8; ++r)
            dec8[mt][r] = __builtin_amdgcn_exp2f(-sdt * LOG2E / fabsf(tp[r]));
    }

    v8f G[4];
#pragma unroll
    for (int mt = 0; mt < 4; ++mt) G[mt] = (v8f){};
    const v8f zero8 = (v8f){};

    const float* xrow = x + (size_t)b * T * F;

#pragma unroll 1
    for (int t = 0; t < T; ++t) {
        // ---- B fragment of [x_t ; 1] (only lanes 0-15 / K=0..6 nonzero) ----
        v16h Bx = {};
        if (hi == 0) {
            const float* xp = xrow + t * F;
            pk2(Bx, 0, xp[0], xp[1]);
            pk2(Bx, 2, xp[2], xp[3]);
            pk2(Bx, 4, xp[4], xp[5]);
            pk2(Bx, 6, 1.0f, 0.0f);
        }

        // u'^T tile (H x 16 batch) = [W_in|bias] x [x;1]^T, A-frag streamed from LDS
        v8f U[4];
#pragma unroll
        for (int mt = 0; mt < 4; ++mt) {
            const uint4 q = *(const uint4*)(lds_ain + (mt * 16 + col) * 4);
            v8u ad;
            ad[0] = hi ? 0u : q.x;   // lanes 16-31 carry K=8..15 -> all zero
            ad[1] = hi ? 0u : q.y;
            ad[2] = hi ? 0u : q.z;
            ad[3] = hi ? 0u : q.w;
            ad[4] = 0u; ad[5] = 0u; ad[6] = 0u; ad[7] = 0u;  // K=16..31 zero
            U[mt] = __builtin_amdgcn_wmma_f32_16x16x32_f16(
                false, __builtin_bit_cast(v16h, ad), false, Bx, (short)0, zero8, false, false);
        }

        // ---- ODE substeps: h^T_new = num + decay*(h^T - num), num = tanh(W_rec h^T + u') ----
        for (int s = 0; s < OS; ++s) {
            v16h B0 = build_bfrag(G[0], G[1], hi);  // K = 0..31  (H rows 0..31)
            v16h B1 = build_bfrag(G[2], G[3], hi);  // K = 32..63
#pragma unroll
            for (int mt = 0; mt < 4; ++mt) {
                v8f acc = __builtin_amdgcn_wmma_f32_16x16x32_f16(
                    false, Arec[mt][0], false, B0, (short)0, U[mt], false, false);
                acc = __builtin_amdgcn_wmma_f32_16x16x32_f16(
                    false, Arec[mt][1], false, B1, (short)0, acc, false, false);
#pragma unroll
                for (int r = 0; r < 8; ++r) {
                    float num = tanh_fast(acc[r]);
                    G[mt][r] = fmaf(dec8[mt][r], G[mt][r] - num, num);
                }
            }
        }

        // ---- fused classifier: logit(b,t) = W_cls . h_b + b_cls (weights from LDS) ----
        float p = 0.0f;
#pragma unroll
        for (int mt = 0; mt < 4; ++mt) {
            const float* wp = lds_wcls + mt * 16 + hi * 8;  // broadcast reads
#pragma unroll
            for (int r = 0; r < 8; ++r) p = fmaf(wp[r], G[mt][r], p);
        }
        // combine the two lane-halves (disjoint H rows) with a VALU cross-lane op
        p += __builtin_bit_cast(float, xhalf(__builtin_bit_cast(unsigned, p)));
        if (hi == 0) out[(size_t)b * T + t] = p + bc;
    }
}

extern "C" void kernel_launch(void* const* d_in, const int* in_sizes, int n_in,
                              void* d_out, int out_size, void* d_ws, size_t ws_size,
                              hipStream_t stream) {
    const float* x    = (const float*)d_in[0];
    const float* dtp  = (const float*)d_in[1];
    const float* Win  = (const float*)d_in[2];
    const float* bin  = (const float*)d_in[3];
    const float* Wrec = (const float*)d_in[4];
    const float* brec = (const float*)d_in[5];
    const float* tau  = (const float*)d_in[6];
    const float* Wcls = (const float*)d_in[7];
    const float* bcls = (const float*)d_in[8];
    const int*   osp  = (const int*)d_in[9];
    float* out = (float*)d_out;

    constexpr int T = 512;
    int Bsz   = out_size / T;          // 4096
    int waves = (Bsz + 15) / 16;       // 256 independent waves (16 batch rows each)
    int block = 256;                   // 8 waves per block, no inter-wave sync in main loop
    int grid  = (waves * 32 + block - 1) / block;

    liquid_gait_kernel<<<grid, block, 0, stream>>>(x, dtp, Win, bin, Wrec, brec,
                                                   tau, Wcls, bcls, osp, out, Bsz);
}
